// Transformer_layer_78228534329470
// MI455X (gfx1250) — matrix operand order repaired and checked by reading, not compiled, not run on hardware
//
#include <hip/hip_runtime.h>
#include <hip/hip_bf16.h>
#include <math.h>

typedef _Float16 h16;
typedef __attribute__((ext_vector_type(16))) _Float16 v16h;
typedef __attribute__((ext_vector_type(8)))  _Float16 v8h;
typedef __attribute__((ext_vector_type(8)))  float    v8f;

#define WMMA_F32_F16(a,b,c) __builtin_amdgcn_wmma_f32_16x16x32_f16(false,(a),false,(b),(short)0,(c),false,false)

static __device__ __forceinline__ v8h ld8(const h16* p) { return *(const v8h*)p; }
static __device__ __forceinline__ v16h comb(v8h lo, v8h hi) {
  v16h r;
#pragma unroll
  for (int i = 0; i < 8; ++i) { r[i] = lo[i]; r[i + 8] = hi[i]; }
  return r;
}

// ---------------------------------------------------------------------------
// f32 -> f16 elementwise convert
// ---------------------------------------------------------------------------
__global__ void cvt_f32_f16_kernel(const float* __restrict__ in, h16* __restrict__ out, int n) {
  int i = blockIdx.x * blockDim.x + threadIdx.x;
  if (i < n) out[i] = (h16)in[i];
}

// ---------------------------------------------------------------------------
// (optional attn-residual add) + LayerNorm.  One row (1024 cols) per block.
// ---------------------------------------------------------------------------
template <bool HAS_ATTN>
__global__ __launch_bounds__(256) void residual_ln_kernel(
    const h16* __restrict__ attn, const float* __restrict__ res,
    const float* __restrict__ g, const float* __restrict__ b,
    float* __restrict__ res_out, h16* __restrict__ norm_out) {
  constexpr int C = 1024;
  const int row = blockIdx.x;
  const int t = threadIdx.x;
  __shared__ float red1[256];
  __shared__ float red2[256];
  float vals[4];
  float s = 0.f, s2 = 0.f;
#pragma unroll
  for (int i = 0; i < 4; ++i) {
    int c = t + i * 256;
    float v = res[(size_t)row * C + c];
    if (HAS_ATTN) {
      v += (float)attn[(size_t)row * C + c];
      res_out[(size_t)row * C + c] = v;
    }
    vals[i] = v;
    s += v;
    s2 += v * v;
  }
  red1[t] = s;
  red2[t] = s2;
  __syncthreads();
  for (int o = 128; o > 0; o >>= 1) {
    if (t < o) { red1[t] += red1[t + o]; red2[t] += red2[t + o]; }
    __syncthreads();
  }
  const float mean = red1[0] * (1.0f / C);
  const float var = red2[0] * (1.0f / C) - mean * mean;
  const float rstd = rsqrtf(var + 1e-5f);
#pragma unroll
  for (int i = 0; i < 4; ++i) {
    int c = t + i * 256;
    norm_out[(size_t)row * C + c] = (h16)((vals[i] - mean) * rstd * g[c] + b[c]);
  }
}

// ---------------------------------------------------------------------------
// Tiled WMMA GEMM: C[MxN] = A[MxK](f16,row) * B[KxN](f16,row)  (+ epilogue)
// 256 threads = 8 waves; block tile 128x128, BK=32; wave tile 64x32.
// Double-buffered LDS, register-staged global loads, one barrier per step.
// ---------------------------------------------------------------------------
enum { EPI_NONE = 0, EPI_BIAS_GELU = 1, EPI_BIAS_RES_F32 = 2 };

template <int EPI>
__global__ __launch_bounds__(256) void gemm_f16_kernel(
    const h16* __restrict__ A, int lda,
    const h16* __restrict__ B, int ldb,
    h16* __restrict__ Ch, float* __restrict__ Cf, int ldc,
    const float* __restrict__ bias,
    const float* __restrict__ resid, int ldr,
    int K) {
  constexpr int BM = 128, BN = 128, BK = 32;
  constexpr int LAS = BK + 8;  // 40 halves
  constexpr int LBS = BK + 8;  // 40 halves (B stored transposed: [n][k])
  __shared__ __align__(16) h16 As[2][BM * LAS];
  __shared__ __align__(16) h16 Bs[2][BN * LBS];

  const int t = threadIdx.x;
  const int wave = t >> 5, lane = t & 31;
  const int lr = lane & 15, ls = lane >> 4;
  const int wm = (wave >> 2) * 64;  // 0,64
  const int wn = (wave & 3) * 32;   // 0,32,64,96
  const int m0 = blockIdx.y * BM;
  const int n0 = blockIdx.x * BN;

  // per-thread staging coordinates (2 chunks of v8h for A and for B)
  const int ar0 = t >> 2, ac0 = (t & 3) * 8;            // chunk0: rows 0..63
  const int bk0 = t >> 4, bn0c = (t & 15) * 8;          // chunk0: k 0..15

  v8f acc[4][2];
#pragma unroll
  for (int im = 0; im < 4; ++im)
#pragma unroll
    for (int in = 0; in < 2; ++in)
#pragma unroll
      for (int r = 0; r < 8; ++r) acc[im][in][r] = 0.f;

  v8h arg[2], brg[2];
  auto gload = [&](int k0) {
    arg[0] = ld8(A + (size_t)(m0 + ar0) * lda + k0 + ac0);
    arg[1] = ld8(A + (size_t)(m0 + 64 + ar0) * lda + k0 + ac0);
    brg[0] = ld8(B + (size_t)(k0 + bk0) * ldb + n0 + bn0c);
    brg[1] = ld8(B + (size_t)(k0 + 16 + bk0) * ldb + n0 + bn0c);
  };
  auto lstore = [&](int buf) {
    *(v8h*)(As[buf] + ar0 * LAS + ac0) = arg[0];
    *(v8h*)(As[buf] + (64 + ar0) * LAS + ac0) = arg[1];
#pragma unroll
    for (int e = 0; e < 8; ++e) Bs[buf][(bn0c + e) * LBS + bk0] = brg[0][e];
#pragma unroll
    for (int e = 0; e < 8; ++e) Bs[buf][(bn0c + e) * LBS + 16 + bk0] = brg[1][e];
  };

  gload(0);
  lstore(0);
  const int ns = K / BK;
  for (int s = 0; s < ns; ++s) {
    __syncthreads();
    const int buf = s & 1;
    if (s + 1 < ns) gload((s + 1) * BK);  // overlap next-tile loads with WMMA

    v16h af[4], bf[2];
#pragma unroll
    for (int im = 0; im < 4; ++im) {
      const h16* p = As[buf] + (wm + im * 16 + lr) * LAS + ls * 8;
      af[im] = comb(ld8(p), ld8(p + 16));  // A: j<8 -> k, j>=8 -> k+16
    }
#pragma unroll
    for (int in = 0; in < 2; ++in) {
      const h16* p = Bs[buf] + (wn + in * 16 + lr) * LBS + ls * 8;
      bf[in] = comb(ld8(p), ld8(p + 16));  // B: j<8 -> k, j>=8 -> k+16
    }
#pragma unroll
    for (int im = 0; im < 4; ++im)
#pragma unroll
      for (int in = 0; in < 2; ++in)
        acc[im][in] = WMMA_F32_F16(af[im], bf[in], acc[im][in]);

    if (s + 1 < ns) lstore(buf ^ 1);
  }

  // epilogue: C element (m,n): row = r + ls*8, col = lane&15 within 16x16 tile
#pragma unroll
  for (int im = 0; im < 4; ++im)
#pragma unroll
    for (int in = 0; in < 2; ++in)
#pragma unroll
      for (int r = 0; r < 8; ++r) {
        const int row = m0 + wm + im * 16 + r + ls * 8;
        const int col = n0 + wn + in * 16 + lr;
        float v = acc[im][in][r];
        if (EPI == EPI_NONE) {
          Ch[(size_t)row * ldc + col] = (h16)v;
        } else if (EPI == EPI_BIAS_GELU) {
          v += bias[col];
          v = 0.5f * v * (1.0f + erff(v * 0.70710678118654752f));
          Ch[(size_t)row * ldc + col] = (h16)v;
        } else {
          v += bias[col] + resid[(size_t)row * ldr + col];
          Cf[(size_t)row * ldc + col] = v;
        }
      }
}

// ---------------------------------------------------------------------------
// Flash attention (wave32 WMMA).  Per (b,h): q,k,v are Nx64 slices of packed
// activations with row strides ldq/ldk/ldv and column offset h*64.
// 128 query rows per block; each wave owns 16 rows; key steps of 32.
// Double-buffered K/V staging; one barrier per key step; wave-private P
// re-layout ordered with s_wait_dscnt only.
// ---------------------------------------------------------------------------
__global__ __launch_bounds__(256) void flash_attn_kernel(
    const h16* __restrict__ Q, int ldq,
    const h16* __restrict__ Kp, int ldk,
    const h16* __restrict__ Vp, int ldv,
    h16* __restrict__ O, int ldo, int N) {
  constexpr int D = 64, BN = 32;
  const int bh = blockIdx.x;
  const int b = bh >> 4;           // HEADS = 16
  const int h = bh & 15;
  const int q0 = blockIdx.y * 128;
  const int t = threadIdx.x, wave = t >> 5, lane = t & 31;
  const int lr = lane & 15, ls = lane >> 4;

  const size_t qbase = (size_t)b * N * ldq + (size_t)h * 64;
  const size_t kbase = (size_t)b * N * ldk + (size_t)h * 64;
  const size_t vbase = (size_t)b * N * ldv + (size_t)h * 64;

  constexpr int KS = D + 8;   // 72 halves
  constexpr int VS = BN + 8;  // 40 halves (Vs stored transposed [d][key])
  __shared__ __align__(16) h16 Ks[2][BN * KS];
  __shared__ __align__(16) h16 VsT[2][D * VS];
  __shared__ __align__(16) h16 Pl[8][16 * 40];  // per-wave 16x32 P scratch

  // resident Q fragments for this wave's 16 rows, pre-scaled by d^-0.5 = 1/8
  const int qrow = q0 + wave * 16 + lr;
  v16h qf[2];
#pragma unroll
  for (int kk = 0; kk < 2; ++kk) {
    const h16* p = Q + qbase + (size_t)qrow * ldq + kk * 32 + ls * 8;
    qf[kk] = comb(ld8(p), ld8(p + 16));
#pragma unroll
    for (int i = 0; i < 16; ++i) qf[kk][i] *= (h16)0.125f;  // exact (2^-3)
  }

  v8f o[4];
#pragma unroll
  for (int dt = 0; dt < 4; ++dt)
#pragma unroll
    for (int r = 0; r < 8; ++r) o[dt][r] = 0.f;
  float mst[8], lst[8];
#pragma unroll
  for (int r = 0; r < 8; ++r) { mst[r] = -1e30f; lst[r] = 0.f; }

  // staging coordinates: 256 threads cover one 32x64 tile in v8h chunks
  const int rr = t >> 3, cc = (t & 7) * 8;
  v8h kreg = ld8(Kp + kbase + (size_t)rr * ldk + cc);
  v8h vreg = ld8(Vp + vbase + (size_t)rr * ldv + cc);
  *(v8h*)(Ks[0] + rr * KS + cc) = kreg;
#pragma unroll
  for (int e = 0; e < 8; ++e) VsT[0][(cc + e) * VS + rr] = vreg[e];

  const int ns = N / BN;
  for (int sIdx = 0; sIdx < ns; ++sIdx) {
    __syncthreads();
    const int buf = sIdx & 1;
    if (sIdx + 1 < ns) {  // overlap next-tile global loads with compute
      const int nn = (sIdx + 1) * BN + rr;
      kreg = ld8(Kp + kbase + (size_t)nn * ldk + cc);
      vreg = ld8(Vp + vbase + (size_t)nn * ldv + cc);
    }

    // scores S[16x32] = (Q*scale)(16x64) x K^T(64x32)
    v8f s[2];
#pragma unroll
    for (int nt = 0; nt < 2; ++nt) {
#pragma unroll
      for (int r = 0; r < 8; ++r) s[nt][r] = 0.f;
#pragma unroll
      for (int ks = 0; ks < 2; ++ks) {
        const h16* p = Ks[buf] + (nt * 16 + lr) * KS + ks * 32 + ls * 8;
        v16h bfrag = comb(ld8(p), ld8(p + 16));
        s[nt] = WMMA_F32_F16(qf[ks], bfrag, s[nt]);
      }
    }

    // online softmax (row = r + ls*8 spread across the 16 lanes of a half-wave)
    float pr[2][8];
#pragma unroll
    for (int r = 0; r < 8; ++r) {
      float s0 = s[0][r], s1 = s[1][r];
      float rowm = fmaxf(s0, s1);
      for (int m = 1; m < 16; m <<= 1) rowm = fmaxf(rowm, __shfl_xor(rowm, m, 32));
      const float newm = fmaxf(mst[r], rowm);
      const float alpha = __expf(mst[r] - newm);
      const float p0 = __expf(s0 - newm);
      const float p1 = __expf(s1 - newm);
      float rs = p0 + p1;
      for (int m = 1; m < 16; m <<= 1) rs += __shfl_xor(rs, m, 32);
      lst[r] = lst[r] * alpha + rs;
      mst[r] = newm;
      pr[0][r] = p0;
      pr[1][r] = p1;
#pragma unroll
      for (int dt = 0; dt < 4; ++dt) o[dt][r] *= alpha;
    }

    // re-layout P: C-fragment -> wave-private LDS -> A-fragment
    h16* pw = &Pl[wave][0];
#pragma unroll
    for (int nt = 0; nt < 2; ++nt)
#pragma unroll
      for (int r = 0; r < 8; ++r)
        pw[(r + ls * 8) * 40 + nt * 16 + lr] = (h16)pr[nt][r];
    __asm__ volatile("s_wait_dscnt 0" ::: "memory");  // wave-local ordering
    v16h pf;
    {
      const h16* p = pw + lr * 40 + ls * 8;
      pf = comb(ld8(p), ld8(p + 16));
    }

    // O(16x64) += P(16x32) x V(32x64)
#pragma unroll
    for (int dt = 0; dt < 4; ++dt) {
      const h16* p = VsT[buf] + (dt * 16 + lr) * VS + ls * 8;
      v16h vf = comb(ld8(p), ld8(p + 16));
      o[dt] = WMMA_F32_F16(pf, vf, o[dt]);
    }

    if (sIdx + 1 < ns) {  // stage next tile into the other buffer
      *(v8h*)(Ks[buf ^ 1] + rr * KS + cc) = kreg;
#pragma unroll
      for (int e = 0; e < 8; ++e) VsT[buf ^ 1][(cc + e) * VS + rr] = vreg[e];
    }
  }

  // normalize and store merged-head output
#pragma unroll
  for (int r = 0; r < 8; ++r) {
    const float inv = 1.0f / lst[r];
    const int row = q0 + wave * 16 + r + ls * 8;
#pragma unroll
    for (int dt = 0; dt < 4; ++dt)
      O[((size_t)b * N + row) * ldo + h * 64 + dt * 16 + lr] = (h16)(o[dt][r] * inv);
  }
}

// ---------------------------------------------------------------------------
// Host launcher
// ---------------------------------------------------------------------------
extern "C" void kernel_launch(void* const* d_in, const int* in_sizes, int n_in,
                              void* d_out, int out_size, void* d_ws, size_t ws_size,
                              hipStream_t stream) {
  (void)in_sizes; (void)n_in; (void)out_size; (void)ws_size;
  constexpr int Bb = 2, N = 2048, M = Bb * N;       // 4096 token rows
  constexpr int DIM = 1024, DF = 384, HID = 1024;

  const float* x      = (const float*)d_in[0];
  const float* feat   = (const float*)d_in[1];
  const float* norm_g = (const float*)d_in[2];
  const float* norm_b = (const float*)d_in[3];
  const float* W_cq   = (const float*)d_in[4];
  const float* W_ckv  = (const float*)d_in[5];
  const float* W_sqkv = (const float*)d_in[6];
  const float* ffn_g  = (const float*)d_in[7];
  const float* ffn_b  = (const float*)d_in[8];
  const float* W1     = (const float*)d_in[9];
  const float* b1     = (const float*)d_in[10];
  const float* W2     = (const float*)d_in[11];
  const float* b2     = (const float*)d_in[12];
  float* out = (float*)d_out;

  char* ws = (char*)d_ws;
  size_t off = 0;
  auto alloc = [&](size_t bytes) -> char* {
    char* p = ws + off;
    off += (bytes + 255) & ~(size_t)255;
    return p;
  };

  h16* wCq   = (h16*)alloc((size_t)DIM * HID * 2);
  h16* wCkv  = (h16*)alloc((size_t)DF * 2 * HID * 2);
  h16* wSqkv = (h16*)alloc((size_t)DIM * 3 * HID * 2);
  h16* w1    = (h16*)alloc((size_t)DIM * 4 * DIM * 2);
  h16* w2    = (h16*)alloc((size_t)4 * DIM * DIM * 2);
  h16* f16f  = (h16*)alloc((size_t)M * DF * 2);
  h16* xn16  = (h16*)alloc((size_t)M * DIM * 2);        // also cn16 / ffn_in16
  h16* q16   = (h16*)alloc((size_t)M * HID * 2);        // q16+kv16+at16 region
  h16* kv16  = (h16*)alloc((size_t)M * 2 * HID * 2);    //   is reused by h1
  h16* at16  = (h16*)alloc((size_t)M * HID * 2);        //   during the FFN
  float* cross = (float*)alloc((size_t)M * DIM * 4);
  h16* qkv16 = (h16*)alloc((size_t)M * 3 * HID * 2);
  h16* h1 = q16;  // 33.5 MB region (q16+kv16+at16) dead by FFN time

  auto cvt = [&](const float* src, h16* dst, int n) {
    cvt_f32_f16_kernel<<<dim3((n + 255) / 256), dim3(256), 0, stream>>>(src, dst, n);
  };
  cvt(W_cq, wCq, DIM * HID);
  cvt(W_ckv, wCkv, DF * 2 * HID);
  cvt(W_sqkv, wSqkv, DIM * 3 * HID);
  cvt(W1, w1, DIM * 4 * DIM);
  cvt(W2, w2, 4 * DIM * DIM);
  cvt(feat, f16f, M * DF);

  // xn = LN(x)
  residual_ln_kernel<false><<<dim3(M), dim3(256), 0, stream>>>(
      nullptr, x, norm_g, norm_b, nullptr, xn16);

  // q = xn @ W_cq
  gemm_f16_kernel<EPI_NONE><<<dim3(HID / 128, M / 128), dim3(256), 0, stream>>>(
      xn16, DIM, wCq, HID, q16, nullptr, HID, nullptr, nullptr, 0, DIM);
  // kv = feature @ W_ckv
  gemm_f16_kernel<EPI_NONE><<<dim3(2 * HID / 128, M / 128), dim3(256), 0, stream>>>(
      f16f, DF, wCkv, 2 * HID, kv16, nullptr, 2 * HID, nullptr, nullptr, 0, DF);
  // cross attention
  flash_attn_kernel<<<dim3(Bb * 16, N / 128), dim3(256), 0, stream>>>(
      q16, HID, kv16, 2 * HID, kv16 + HID, 2 * HID, at16, HID, N);
  // cross = attn + x ; cn = LN(cross)
  residual_ln_kernel<true><<<dim3(M), dim3(256), 0, stream>>>(
      at16, x, norm_g, norm_b, cross, xn16);

  // qkv = cn @ W_sqkv
  gemm_f16_kernel<EPI_NONE><<<dim3(3 * HID / 128, M / 128), dim3(256), 0, stream>>>(
      xn16, DIM, wSqkv, 3 * HID, qkv16, nullptr, 3 * HID, nullptr, nullptr, 0, DIM);
  // self attention
  flash_attn_kernel<<<dim3(Bb * 16, N / 128), dim3(256), 0, stream>>>(
      qkv16, 3 * HID, qkv16 + HID, 3 * HID, qkv16 + 2 * HID, 3 * HID, at16, HID, N);
  // self_tok = attn + cross (in place) ; ffn_in = LN(self_tok, ffn_g, ffn_b)
  residual_ln_kernel<true><<<dim3(M), dim3(256), 0, stream>>>(
      at16, cross, ffn_g, ffn_b, cross, xn16);

  // h1 = gelu(ffn_in @ W1 + b1)
  gemm_f16_kernel<EPI_BIAS_GELU><<<dim3(4 * DIM / 128, M / 128), dim3(256), 0, stream>>>(
      xn16, DIM, w1, 4 * DIM, h1, nullptr, 4 * DIM, b1, nullptr, 0, DIM);
  // out = h1 @ W2 + b2 + self_tok
  gemm_f16_kernel<EPI_BIAS_RES_F32><<<dim3(DIM / 128, M / 128), dim3(256), 0, stream>>>(
      h1, 4 * DIM, w2, DIM, nullptr, out, DIM, b2, cross, DIM, 4 * DIM);
}
